// GridSampleConv_14920716386524
// MI455X (gfx1250) — matrix operand order repaired and checked by reading, not compiled, not run on hardware
//
#include <hip/hip_runtime.h>

// Problem constants (match reference setup_inputs)
#define BATCH 4
#define NPTS  16384
#define CIN   64
#define COUT  128
#define KNNK  16
#define KPTS  27
#define KCH   (KPTS * CIN)          // 1728 = GEMM K dimension
#define KP_EXTENT_INV 4.0f          // 1 / (0.5/2)

typedef __attribute__((ext_vector_type(16))) _Float16 v16h;
typedef __attribute__((ext_vector_type(8)))  _Float16 v8h;
typedef __attribute__((ext_vector_type(4)))  _Float16 v4h;
typedef __attribute__((ext_vector_type(8)))  float    v8f;

// ---------------------------------------------------------------------------
// Kernel 1: brute-force KNN (top-16 by squared distance, self included).
// One thread per query point; candidates staged through LDS in 256-point
// tiles; sorted best-16 kept in registers via fully-unrolled bubble insert
// (static indices -> no scratch spill).
// ---------------------------------------------------------------------------
__global__ __launch_bounds__(256) void knn_kernel(const float* __restrict__ points,
                                                  int* __restrict__ knn_idx) {
    const int b = blockIdx.y;
    const int n = blockIdx.x * 256 + threadIdx.x;
    const float* pb = points + (size_t)b * NPTS * 3;

    const float qx = pb[n * 3 + 0];
    const float qy = pb[n * 3 + 1];
    const float qz = pb[n * 3 + 2];

    __shared__ float4 tile[256];

    float bd[KNNK];
    int   bi[KNNK];
#pragma unroll
    for (int i = 0; i < KNNK; ++i) { bd[i] = 3.4e38f; bi[i] = 0; }

    for (int j0 = 0; j0 < NPTS; j0 += 256) {
        const int c = j0 + threadIdx.x;
        tile[threadIdx.x] = make_float4(pb[c * 3 + 0], pb[c * 3 + 1], pb[c * 3 + 2], 0.0f);
        __syncthreads();
#pragma unroll 4
        for (int j = 0; j < 256; ++j) {
            const float4 p = tile[j];
            const float dx = p.x - qx, dy = p.y - qy, dz = p.z - qz;
            const float d = dx * dx + dy * dy + dz * dz;
            if (d < bd[KNNK - 1]) {
                bd[KNNK - 1] = d; bi[KNNK - 1] = j0 + j;
#pragma unroll
                for (int s = KNNK - 1; s > 0; --s) {
                    if (bd[s] < bd[s - 1]) {
                        const float td = bd[s]; bd[s] = bd[s - 1]; bd[s - 1] = td;
                        const int   ti = bi[s]; bi[s] = bi[s - 1]; bi[s - 1] = ti;
                    }
                }
            }
        }
        __syncthreads();
    }

    int* o = knn_idx + ((size_t)b * NPTS + n) * KNNK;
#pragma unroll
    for (int i = 0; i < KNNK; ++i) o[i] = bi[i];
}

// ---------------------------------------------------------------------------
// Kernel 2: weights [27][64][128] fp32 -> transposed f16 Wt[128][1728] so the
// WMMA B-operand (16 consecutive K values per lane) is one contiguous 32B load.
// ---------------------------------------------------------------------------
__global__ __launch_bounds__(256) void wtranspose_kernel(const float* __restrict__ W,
                                                         _Float16* __restrict__ Wt) {
    const int gid = blockIdx.x * 256 + threadIdx.x;
    if (gid >= COUT * KCH) return;
    const int o  = gid / KCH;
    const int kc = gid - o * KCH;          // kc = p*64 + c
    Wt[gid] = (_Float16)W[(size_t)kc * COUT + o];
}

// ---------------------------------------------------------------------------
// Kernel 3: fused KPConv. One block = one 16-point M-tile.
//   Phase A: gather neighbor features (f16) + kernel-point influences (f16).
//   Per kernel point p: build wfeat chunk A[16][64] (f16) in LDS, then the
//   8 waves run v_wmma_f32_16x16x32_f16 (2 k-steps) each for its 16-wide
//   Cout slice, accumulating in v8f across all 27 chunks.
// ---------------------------------------------------------------------------
__global__ __launch_bounds__(256) void kpconv_kernel(const float* __restrict__ points,
                                                     const float* __restrict__ features,
                                                     const float* __restrict__ kps,
                                                     const _Float16* __restrict__ Wt,
                                                     const int* __restrict__ knn_idx,
                                                     float* __restrict__ out) {
    __shared__ _Float16 sF[16][16][64];     // neighbor features (f16)       32 KB
    __shared__ _Float16 sInfl[16][16][28];  // influences, padded 27->28     14 KB
    __shared__ _Float16 sA[16][72];         // wfeat chunk (WMMA A), padded   2.3 KB
    __shared__ int      sIdx[16][16];
    __shared__ float    sQ[16][3];
    __shared__ float    sKP[27][3];

    const int tid = threadIdx.x;
    const int bid = blockIdx.x;
    const int b   = bid / (NPTS / 16);
    const int n0  = (bid % (NPTS / 16)) * 16;

    // ---- Phase A0: tile metadata ----
    const int m = tid >> 4;
    const int k = tid & 15;
    sIdx[m][k] = knn_idx[((size_t)b * NPTS + n0 + m) * KNNK + k];
    if (tid < 48) sQ[tid / 3][tid % 3] = points[((size_t)b * NPTS + n0 + tid / 3) * 3 + tid % 3];
    if (tid < 81) sKP[tid / 3][tid % 3] = kps[tid];
    __syncthreads();

    // ---- Phase A1: per (m,k): influences + feature gather ----
    {
        const int nbr = sIdx[m][k];
        const float* pp = points + ((size_t)b * NPTS + nbr) * 3;
        const float rx = pp[0] - sQ[m][0];
        const float ry = pp[1] - sQ[m][1];
        const float rz = pp[2] - sQ[m][2];
#pragma unroll 9
        for (int p = 0; p < KPTS; ++p) {
            const float dx = rx - sKP[p][0];
            const float dy = ry - sKP[p][1];
            const float dz = rz - sKP[p][2];
            const float dist = sqrtf(dx * dx + dy * dy + dz * dz + 1e-12f);
            sInfl[m][k][p] = (_Float16)fmaxf(0.0f, 1.0f - dist * KP_EXTENT_INV);
        }
        const float4* frow = (const float4*)(features + ((size_t)b * NPTS + nbr) * CIN);
#pragma unroll
        for (int c4 = 0; c4 < CIN / 4; ++c4) {
            const float4 f = frow[c4];
            sF[m][k][c4 * 4 + 0] = (_Float16)f.x;
            sF[m][k][c4 * 4 + 1] = (_Float16)f.y;
            sF[m][k][c4 * 4 + 2] = (_Float16)f.z;
            sF[m][k][c4 * 4 + 3] = (_Float16)f.w;
        }
    }
    __syncthreads();

    // ---- Phase B: 27 x (aggregate chunk -> WMMA) ----
    const int w    = tid >> 5;        // wave id: Cout tile [16w, 16w+16)
    const int lane = tid & 31;
    const int nloc = lane & 15;       // A-row / B-col / D-col within tile
    const int hi   = lane >> 4;       // K half-select per ISA lane layout
    const int c0   = (tid & 15) * 4;  // phase-B aggregation mapping

    v8f acc = {};
    const _Float16* wbase = Wt + (size_t)(w * 16 + nloc) * KCH;

    for (int p = 0; p < KPTS; ++p) {
        // wfeat[m][c] = sum_k infl[m][k][p] * F[m][k][c]  (4 c-values/thread)
        float a0 = 0.f, a1 = 0.f, a2 = 0.f, a3 = 0.f;
#pragma unroll
        for (int kk = 0; kk < KNNK; ++kk) {
            const float wk = (float)sInfl[m][kk][p];
            const v4h f = *(const v4h*)&sF[m][kk][c0];
            a0 += wk * (float)f[0];
            a1 += wk * (float)f[1];
            a2 += wk * (float)f[2];
            a3 += wk * (float)f[3];
        }
        v4h av;
        av[0] = (_Float16)a0; av[1] = (_Float16)a1;
        av[2] = (_Float16)a2; av[3] = (_Float16)a3;
        *(v4h*)&sA[m][c0] = av;
        __syncthreads();

        // Two 16x16x32 WMMA k-steps cover this p's 64-wide K slice.
        const _Float16* wp = wbase + p * CIN;
#pragma unroll
        for (int ks = 0; ks < 64; ks += 32) {
            // A fragment: 16 contiguous K halves per lane (matches B)
            const v8h alo = *(const v8h*)&sA[nloc][ks + hi * 16];
            const v8h ahi = *(const v8h*)&sA[nloc][ks + hi * 16 + 8];
            v16h afrag;
#pragma unroll
            for (int i = 0; i < 8; ++i) { afrag[i] = alo[i]; afrag[i + 8] = ahi[i]; }
            // B fragment: 16 contiguous K halves for this lane's column
            const v16h bfrag = *(const v16h*)(wp + ks + hi * 16);
            acc = __builtin_amdgcn_wmma_f32_16x16x32_f16(
                false, afrag, false, bfrag, (short)0, acc, false, false);
        }
        __syncthreads();
    }

    // ---- Store: D layout — VGPR r holds row r + 8*hi, col = nloc ----
    float* obase = out + ((size_t)b * NPTS + n0) * COUT + w * 16 + nloc;
#pragma unroll
    for (int r = 0; r < 8; ++r) obase[(size_t)(r + hi * 8) * COUT] = acc[r];
}

// ---------------------------------------------------------------------------
extern "C" void kernel_launch(void* const* d_in, const int* in_sizes, int n_in,
                              void* d_out, int out_size, void* d_ws, size_t ws_size,
                              hipStream_t stream) {
    (void)in_sizes; (void)n_in; (void)out_size; (void)ws_size;
    const float* points   = (const float*)d_in[0];  // [B,N,3]
    const float* features = (const float*)d_in[1];  // [B,N,64]
    const float* kps      = (const float*)d_in[2];  // [27,3]
    const float* weights  = (const float*)d_in[3];  // [27,64,128]
    float* out = (float*)d_out;                     // [B,N,128] fp32

    int*      knn_idx = (int*)d_ws;                                        // 4 MB
    _Float16* Wt      = (_Float16*)((char*)d_ws +
                          (size_t)BATCH * NPTS * KNNK * sizeof(int));      // 432 KB

    knn_kernel<<<dim3(NPTS / 256, BATCH), 256, 0, stream>>>(points, knn_idx);
    wtranspose_kernel<<<(COUT * KCH + 255) / 256, 256, 0, stream>>>(weights, Wt);
    kpconv_kernel<<<BATCH * (NPTS / 16), 256, 0, stream>>>(points, features, kps,
                                                           Wt, knn_idx, out);
}
